// Embedding_56908316672257
// MI455X (gfx1250) — matrix operand order repaired and checked by reading, not compiled, not run on hardware
//
#include <hip/hip_runtime.h>
#include <hip/hip_bf16.h>
#include <math.h>

typedef __attribute__((ext_vector_type(16))) _Float16 v16h;
typedef __attribute__((ext_vector_type(8)))  float    v8f;

// ---------------------------------------------------------------------------
// WMMA helpers (CDNA5 / gfx1250, wave32)
// ---------------------------------------------------------------------------
static __device__ __forceinline__ v8f wmma_f16(v16h a, v16h b, v8f c) {
    // D = A(16x32 f16) * B(32x16 f16) + C(16x16 f32)
    return __builtin_amdgcn_wmma_f32_16x16x32_f16(
        /*neg_a=*/false, a, /*neg_b=*/false, b,
        /*c_mod=*/(short)0, c, /*reuse_a=*/false, /*reuse_b=*/false);
}

// A-matrix 16x32 f16 layout (ISA 7.12.2):
//   lane L: row = L&15; kb = (L<16)?0:8
//   a[j]   = A[row][kb+j]       j=0..7   (VGPRs 0..3)
//   a[8+j] = A[row][16+kb+j]    j=0..7   (VGPRs 4..7)
// KMAX is compile-time; out-of-range K reads are address-clamped + value
// selected to zero (unconditional loads, no EXEC branching).
template <int KMAX>
static __device__ __forceinline__ v16h load_A_row(const float* __restrict__ rowptr) {
    const int lane = threadIdx.x & 31;
    const int kb = (lane < 16) ? 0 : 8;
    v16h a;
#pragma unroll
    for (int j = 0; j < 8; ++j) {
        int k0 = kb + j;
        int k1 = 16 + kb + j;
        if (KMAX >= 32) {
            a[j]     = (_Float16)rowptr[k0];
            a[8 + j] = (_Float16)rowptr[k1];
        } else {
            int k0c = (k0 < KMAX) ? k0 : 0;
            int k1c = (k1 < KMAX) ? k1 : 0;
            float v0 = rowptr[k0c];     // unconditional load, clamped addr
            float v1 = rowptr[k1c];
            a[j]     = (_Float16)((k0 < KMAX) ? v0 : 0.0f);   // v_cndmask
            a[8 + j] = (_Float16)((k1 < KMAX) ? v1 : 0.0f);
        }
    }
    return a;
}

// B-matrix 32x16 f16 layout (ISA 7.12.2/7.12.4 pattern):
//   lane L: col n = ncol0 + (L&15); kb = (L<16)?0:8
//   b[j] = B[kb+j][n], b[8+j] = B[16+kb+j][n]   j=0..7
// W is row-major [KMAX x ldw] f32; k >= KMAX zero-padded (clamp+select).
template <int KMAX>
static __device__ __forceinline__ v16h load_B_w(const float* __restrict__ W,
                                                int ldw, int ncol0) {
    const int lane = threadIdx.x & 31;
    const int n = ncol0 + (lane & 15);
    const int kb = (lane < 16) ? 0 : 8;
    v16h b;
#pragma unroll
    for (int v = 0; v < 8; ++v) {
        int k0 = kb + ((2 * v) & 7) + ((v >= 4) ? 16 : 0);
        int k1 = k0 + 1;
        if (KMAX >= 32) {
            b[2 * v]     = (_Float16)W[(size_t)k0 * ldw + n];
            b[2 * v + 1] = (_Float16)W[(size_t)k1 * ldw + n];
        } else {
            int k0c = (k0 < KMAX) ? k0 : 0;
            int k1c = (k1 < KMAX) ? k1 : 0;
            float v0 = W[(size_t)k0c * ldw + n];
            float v1 = W[(size_t)k1c * ldw + n];
            b[2 * v]     = (_Float16)((k0 < KMAX) ? v0 : 0.0f);
            b[2 * v + 1] = (_Float16)((k1 < KMAX) ? v1 : 0.0f);
        }
    }
    return b;
}

// Same-wave LDS producer->consumer fence: DS ops are in-order per wave; the
// split-counter wait + scheduling barriers keep the compiler honest.
static __device__ __forceinline__ void wave_lds_fence() {
    __builtin_amdgcn_wave_barrier();
    asm volatile("s_wait_dscnt 0" ::: "memory");
    __builtin_amdgcn_wave_barrier();
}

// silu via v_rcp_f32 (TRANS), avoiding the correctly-rounded divide sequence.
static __device__ __forceinline__ float silu(float v) {
    return v * __builtin_amdgcn_rcpf(1.0f + __expf(-v));
}

// ---------------------------------------------------------------------------
// Kernel 1: per-edge preprocessing
//   edge_vec -> spherical harmonics (lmax=2), length -> cosine basis
//   writes eaf [E,17], emb [E,10], and copies src/dst indices to output.
// ---------------------------------------------------------------------------
__global__ void __launch_bounds__(256)
edge_pre_kernel(const float* __restrict__ pos,
                const float* __restrict__ edge_attr,
                const int* __restrict__ eidx,
                float* __restrict__ eaf, float* __restrict__ emb,
                int* __restrict__ out_src, int* __restrict__ out_dst, int E) {
    int e = blockIdx.x * blockDim.x + threadIdx.x;
    if (e >= E) return;
    int s = eidx[e];
    int d = eidx[(size_t)E + e];
    out_src[e] = s;
    out_dst[e] = d;

    float vx = pos[3 * (size_t)s + 0] - pos[3 * (size_t)d + 0];
    float vy = pos[3 * (size_t)s + 1] - pos[3 * (size_t)d + 1];
    float vz = pos[3 * (size_t)s + 2] - pos[3 * (size_t)d + 2];
    float r = sqrtf(vx * vx + vy * vy + vz * vz);
    float inv = __builtin_amdgcn_rcpf(fmaxf(r, 1e-9f));
    float x = vx * inv, y = vy * inv, z = vz * inv;

    const float s3 = 1.73205080756887729f;   // sqrt(3)
    const float s15 = 3.87298334620741689f;  // sqrt(15)
    const float s5 = 2.23606797749978969f;   // sqrt(5)
    float sh[9];
    sh[0] = 1.0f;
    sh[1] = s3 * x; sh[2] = s3 * y; sh[3] = s3 * z;
    sh[4] = s15 * x * y;
    sh[5] = s15 * y * z;
    sh[6] = 0.5f * s5 * (3.0f * z * z - 1.0f);
    sh[7] = s15 * x * z;
    sh[8] = 0.5f * s15 * (x * x - y * y);

    float* o = eaf + (size_t)e * 17;
#pragma unroll
    for (int j = 0; j < 8; ++j) o[j] = edge_attr[(size_t)e * 8 + j];
#pragma unroll
    for (int j = 0; j < 9; ++j) o[8 + j] = sh[j];

    // cosine basis: values_i = (i+1)*4/11, step = 4/11, * sqrt(10)
    const float step = 4.0f / 11.0f;
    const float istep = 11.0f / 4.0f;
    const float sq10 = 3.16227766016837933f;
    const float hpi = 1.57079632679489662f;
    float* m = emb + (size_t)e * 10;
#pragma unroll
    for (int i = 0; i < 10; ++i) {
        float diff = (r - (float)(i + 1) * step) * istep;
        m[i] = (diff > -1.0f && diff < 1.0f) ? __cosf(hpi * diff) * sq10 : 0.0f;
    }
}

// ---------------------------------------------------------------------------
// Kernel 2: per-layer edge message kernel. One wave handles 16 edges/tile:
//   P = h[src]@Ws ; Q = eaf@Wa ; w = silu(emb@f1)@f2 ; scatter P*w+Q -> agg
// 8 v_wmma_f32_16x16x32_f16 per tile.
// ---------------------------------------------------------------------------
template <int DIN>
__global__ void __launch_bounds__(256)
edge_layer_kernel(const float* __restrict__ h_in,
                  const float* __restrict__ Ws, const float* __restrict__ Wa,
                  const float* __restrict__ f1, const float* __restrict__ f2,
                  const float* __restrict__ eaf, const float* __restrict__ emb,
                  const int* __restrict__ src, const int* __restrict__ dst,
                  float* __restrict__ agg, int E) {
    __shared__ float swf[8][16 * 32];   // per-wave w-restage (C-layout -> A-layout)

    const int lane = threadIdx.x & 31;
    const int wib = threadIdx.x >> 5;
    const int wid = blockIdx.x * (blockDim.x >> 5) + wib;
    const int nw = gridDim.x * (blockDim.x >> 5);
    const int ntiles = (E + 15) >> 4;

    // Weight B-tiles: loaded once per wave, L2-resident broadcasts.
    const v16h bs0 = load_B_w<DIN>(Ws, 32, 0), bs1 = load_B_w<DIN>(Ws, 32, 16);
    const v16h ba0 = load_B_w<17>(Wa, 32, 0),  ba1 = load_B_w<17>(Wa, 32, 16);
    const v16h b10 = load_B_w<10>(f1, 32, 0),  b11 = load_B_w<10>(f1, 32, 16);
    const v16h b20 = load_B_w<32>(f2, 32, 0),  b21 = load_B_w<32>(f2, 32, 16);

    const int col = lane & 15;
    const int ro = (lane < 16) ? 0 : 8;
    float* myswf = &swf[wib][0];

    for (int t = wid; t < ntiles; t += nw) {
        const int tb = t << 4;
        const bool full = (tb + 16 <= E);
        int e_row = tb + (lane & 15);
        int e_cl = (e_row < E) ? e_row : (E - 1);

        // Destination rows for this lane's 8 output rows: contiguous in dst[].
        int drow[8];
        if (full) {
#pragma unroll
            for (int r = 0; r < 8; ++r) drow[r] = dst[tb + ro + r];
        } else {
#pragma unroll
            for (int r = 0; r < 8; ++r) {
                int i = tb + ro + r;
                drow[r] = dst[(i < E) ? i : (E - 1)];
            }
        }

        // P = h[src] @ Ws
        int srow = src[e_cl];
        v16h ah = load_A_row<DIN>(h_in + (size_t)srow * DIN);
        v8f p0 = {}, p1 = {};
        p0 = wmma_f16(ah, bs0, p0);
        p1 = wmma_f16(ah, bs1, p1);

        // Q = eaf @ Wa
        v16h ae = load_A_row<17>(eaf + (size_t)e_cl * 17);
        v8f q0 = {}, q1 = {};
        q0 = wmma_f16(ae, ba0, q0);
        q1 = wmma_f16(ae, ba1, q1);

        // t = emb @ f1  -> silu -> restage via LDS -> w = . @ f2
        v16h am = load_A_row<10>(emb + (size_t)e_cl * 10);
        v8f t0 = {}, t1 = {};
        t0 = wmma_f16(am, b10, t0);
        t1 = wmma_f16(am, b11, t1);
#pragma unroll
        for (int r = 0; r < 8; ++r) {
            myswf[(r + ro) * 32 + col]      = silu(t0[r]);
            myswf[(r + ro) * 32 + 16 + col] = silu(t1[r]);
        }
        wave_lds_fence();

        v16h aw = load_A_row<32>(myswf + (lane & 15) * 32);
        v8f w0 = {}, w1 = {};
        w0 = wmma_f16(aw, b20, w0);
        w1 = wmma_f16(aw, b21, w1);

        // m = P*w + Q ; scatter into agg[dst]
        if (full) {
#pragma unroll
            for (int r = 0; r < 8; ++r) {
                atomicAdd(&agg[(size_t)drow[r] * 32 + col],      p0[r] * w0[r] + q0[r]);
                atomicAdd(&agg[(size_t)drow[r] * 32 + 16 + col], p1[r] * w1[r] + q1[r]);
            }
        } else {
#pragma unroll
            for (int r = 0; r < 8; ++r) {
                if (tb + ro + r < E) {
                    atomicAdd(&agg[(size_t)drow[r] * 32 + col],      p0[r] * w0[r] + q0[r]);
                    atomicAdd(&agg[(size_t)drow[r] * 32 + 16 + col], p1[r] * w1[r] + q1[r]);
                }
            }
        }
        // DS ops are serviced in order per wave; a scheduling-only barrier is
        // enough to keep next iteration's LDS stores behind this one's loads.
        __builtin_amdgcn_wave_barrier();
    }
}

// ---------------------------------------------------------------------------
// Kernel 3: per-layer node update:  h_out = h@Wf + agg/sqrt(32)  (+silu)
// ---------------------------------------------------------------------------
template <int DIN>
__global__ void __launch_bounds__(256)
node_layer_kernel(const float* __restrict__ h_in,
                  const float* __restrict__ Wf,
                  const float* __restrict__ agg,
                  float* __restrict__ h_out, int N, int do_silu) {
    const int lane = threadIdx.x & 31;
    const int wib = threadIdx.x >> 5;
    const int wid = blockIdx.x * (blockDim.x >> 5) + wib;
    const int nw = gridDim.x * (blockDim.x >> 5);
    const int ntiles = (N + 15) >> 4;
    const float rsq = 0.17677669529663689f;  // 1/sqrt(32)

    const v16h bf0 = load_B_w<DIN>(Wf, 32, 0), bf1 = load_B_w<DIN>(Wf, 32, 16);
    const int col = lane & 15;
    const int ro = (lane < 16) ? 0 : 8;

    for (int t = wid; t < ntiles; t += nw) {
        const int tb = t << 4;
        int n_row = tb + (lane & 15);
        int n_cl = (n_row < N) ? n_row : (N - 1);
        v16h a = load_A_row<DIN>(h_in + (size_t)n_cl * DIN);
        v8f c0 = {}, c1 = {};
        c0 = wmma_f16(a, bf0, c0);
        c1 = wmma_f16(a, bf1, c1);
#pragma unroll
        for (int r = 0; r < 8; ++r) {
            int n = tb + r + ro;
            if (n < N) {
                float o0 = c0[r] + agg[(size_t)n * 32 + col] * rsq;
                float o1 = c1[r] + agg[(size_t)n * 32 + 16 + col] * rsq;
                if (do_silu) { o0 = silu(o0); o1 = silu(o1); }
                h_out[(size_t)n * 32 + col] = o0;
                h_out[(size_t)n * 32 + 16 + col] = o1;
            }
        }
    }
}

// ---------------------------------------------------------------------------
// Host launch
// ---------------------------------------------------------------------------
extern "C" void kernel_launch(void* const* d_in, const int* in_sizes, int n_in,
                              void* d_out, int out_size, void* d_ws, size_t ws_size,
                              hipStream_t stream) {
    const float* pos   = (const float*)d_in[0];
    const float* x     = (const float*)d_in[1];
    const float* nattr = (const float*)d_in[2];
    const float* eattr = (const float*)d_in[3];
    const int*   eidx  = (const int*)d_in[4];
    const int*   batch = (const int*)d_in[5];
    const float* Ws[3] = {(const float*)d_in[6],  (const float*)d_in[11], (const float*)d_in[16]};
    const float* Wa[3] = {(const float*)d_in[7],  (const float*)d_in[12], (const float*)d_in[17]};
    const float* F1[3] = {(const float*)d_in[8],  (const float*)d_in[13], (const float*)d_in[18]};
    const float* F2[3] = {(const float*)d_in[9],  (const float*)d_in[14], (const float*)d_in[19]};
    const float* Wf[3] = {(const float*)d_in[10], (const float*)d_in[15], (const float*)d_in[20]};

    const int N = in_sizes[0] / 3;   // pos is [N,3]
    const int E = in_sizes[3] / 8;   // edge_attr is [E,8]

    // d_out tuple layout (flat, return order):
    float* out = (float*)d_out;
    size_t off = 0;
    float* out_h   = out + off;            off += (size_t)N * 32;
    float* out_na  = out + off;            off += (size_t)N * 8;
    int*   out_src = (int*)(out + off);    off += (size_t)E;
    int*   out_dst = (int*)(out + off);    off += (size_t)E;
    float* out_eaf = out + off;            off += (size_t)E * 17;
    float* out_emb = out + off;            off += (size_t)E * 10;
    int*   out_bat = (int*)(out + off);

    // workspace: agg [N,32] + two h ping-pong buffers [N,32]
    float* agg = (float*)d_ws;
    float* hA  = agg + (size_t)N * 32;
    float* hB  = hA + (size_t)N * 32;

    hipMemcpyAsync(out_na, nattr, (size_t)N * 8 * sizeof(float),
                   hipMemcpyDeviceToDevice, stream);
    hipMemcpyAsync(out_bat, batch, (size_t)N * sizeof(int),
                   hipMemcpyDeviceToDevice, stream);

    edge_pre_kernel<<<(E + 255) / 256, 256, 0, stream>>>(
        pos, eattr, eidx, out_eaf, out_emb, out_src, out_dst, E);

    // Layer 0: din = 16 (input x), layers 1-2: din = 32
    hipMemsetAsync(agg, 0, (size_t)N * 32 * sizeof(float), stream);
    edge_layer_kernel<16><<<2048, 256, 0, stream>>>(
        x, Ws[0], Wa[0], F1[0], F2[0], out_eaf, out_emb, out_src, out_dst, agg, E);
    node_layer_kernel<16><<<256, 256, 0, stream>>>(
        x, Wf[0], agg, hA, N, 1);

    hipMemsetAsync(agg, 0, (size_t)N * 32 * sizeof(float), stream);
    edge_layer_kernel<32><<<2048, 256, 0, stream>>>(
        hA, Ws[1], Wa[1], F1[1], F2[1], out_eaf, out_emb, out_src, out_dst, agg, E);
    node_layer_kernel<32><<<256, 256, 0, stream>>>(
        hA, Wf[1], agg, hB, N, 1);

    hipMemsetAsync(agg, 0, (size_t)N * 32 * sizeof(float), stream);
    edge_layer_kernel<32><<<2048, 256, 0, stream>>>(
        hB, Ws[2], Wa[2], F1[2], F2[2], out_eaf, out_emb, out_src, out_dst, agg, E);
    node_layer_kernel<32><<<256, 256, 0, stream>>>(
        hB, Wf[2], agg, out_h, N, 0);
}
